// StandardVQ_27779848471400
// MI455X (gfx1250) — compile-verified, matrix operand order checked by reading, not run on hardware
//
#include <hip/hip_runtime.h>
#include <hip/hip_bf16.h>
#include <math.h>

// ---------------------------------------------------------------------------
// Problem dims (fixed by reference)
// ---------------------------------------------------------------------------
#define NROWS 16384
#define INDIM 512
#define HID   2048
#define LAT   256
#define CODES 8192

typedef __bf16 bf16;
typedef __attribute__((ext_vector_type(16))) __bf16 v16bf;
typedef __attribute__((ext_vector_type(8)))  __bf16 bf16x8;
typedef __attribute__((ext_vector_type(8)))  float  v8f;

// ---------------------------------------------------------------------------
// CDNA5 async copy (global -> LDS, tracked by ASYNCcnt) + waits.
// VGLOBAL async encoding: VDST = LDS byte-address VGPR, VADDR = 64-bit global
// address VGPR pair, SADDR = off.  Generic "shared" pointers on amdgcn are
// {aperture_hi32, lds_offset32}, so truncating to 32 bits yields the DS addr.
// ---------------------------------------------------------------------------
__device__ __forceinline__ void async_copy_b128(unsigned lds_addr, const void* gaddr) {
    asm volatile("global_load_async_to_lds_b128 %0, %1, off"
                 :: "v"(lds_addr), "v"(gaddr)
                 : "memory");
}
template<int N>
__device__ __forceinline__ void async_wait() {
    asm volatile("s_wait_asynccnt %0" :: "i"(N) : "memory");
}

__device__ __forceinline__ v8f wmma_bf16(v16bf a, v16bf b, v8f c) {
    return __builtin_amdgcn_wmma_f32_16x16x32_bf16(
        /*neg_a=*/false, a, /*neg_b=*/false, b,
        /*c_mod=*/(short)0, c, /*reuse_a=*/false, /*reuse_b=*/false);
}

__device__ __forceinline__ float gelu_exact(float v) {
    return 0.5f * v * (1.0f + erff(v * 0.70710678118654752440f));
}

// ---------------------------------------------------------------------------
// 3-stage ring-buffered LDS WMMA GEMM core.
//   A  [M][K] bf16 row-major,  Bt [N][K] bf16 row-major (transposed weight)
//   Block = 256 threads = 8 waves (2 M-waves x 4 N-waves)
//   Block tile 64x256, wave tile 32x64, K-step 32.
// LDS tiles live in *fragment order*: per 16x32 subtile, per 16-byte chunk c
// (c=0 -> K=(l>>4)*8.., c=1 -> K=16+(l>>4)*8..), per lane l, one bf16x8.
// Fragment ds_load_b128 reads are stride-1 across lanes (bank-conflict free);
// the async staging addresses apply the layout permutation for free.
//   per-stage buffer: A 4 KB + B 16 KB = 20 KB;  ring = 3 x 20 KB = 60 KB.
// Pipeline: each stage issues exactly 5 async instructions per wave and
// ASYNCcnt retires in order, so  s_wait_asynccnt <= 5  ==  "stage s done,
// stage s+1 may remain in flight".  Two tile-DMAs stay in flight behind
// every compute step.
// ---------------------------------------------------------------------------
#define TILE_BYTES 20480
#define B_OFF_B128 256   // B region starts 4096 B into a buffer = index 256 in b128s

__device__ __forceinline__ void gemm_core_lds(
    const bf16* __restrict__ A, const bf16* __restrict__ Bt, int K,
    int m0blk, int n0blk, char* smem, v8f acc[2][4])
{
    const int t    = threadIdx.x;
    const int wave = t >> 5;
    const int wm   = wave >> 2;   // 0..1
    const int wn   = wave & 3;    // 0..3
    const int lane = t & 31;
    const int NK   = K >> 5;

    const unsigned lds_base = (unsigned)(unsigned long long)smem;

    // --- per-thread staging assignment (1 A-b128 + 4 B-b128 per K-step) ---
    const int sA = t >> 6;            // A subtile 0..3
    const int cA = (t >> 5) & 1;      // chunk
    const int lA = t & 31;            // lane slot
    const size_t gA = (size_t)(m0blk + sA * 16 + (lA & 15)) * K
                    + ((lA >> 4) << 3) + cA * 16;
    const unsigned oA = (unsigned)(((sA * 2 + cA) * 32 + lA) * 16);

    size_t   gB[4];
    unsigned oB[4];
#pragma unroll
    for (int q = 0; q < 4; ++q) {
        const int idx = t + q * 256;
        const int sB = idx >> 6;          // 0..15
        const int cB = (idx >> 5) & 1;
        const int lB = idx & 31;
        gB[q] = (size_t)(n0blk + sB * 16 + (lB & 15)) * K
              + ((lB >> 4) << 3) + cB * 16;
        oB[q] = (unsigned)(4096 + ((sB * 2 + cB) * 32 + lB) * 16);
    }

    auto stage = [&](int buf, int k0) {
        const unsigned bb = lds_base + buf * TILE_BYTES;
        async_copy_b128(bb + oA, (const void*)(A + gA + k0));
#pragma unroll
        for (int q = 0; q < 4; ++q)
            async_copy_b128(bb + oB[q], (const void*)(Bt + gB[q] + k0));
    };

    auto compute = [&](int s) {
        const bf16x8* P = (const bf16x8*)(smem + (s % 3) * TILE_BYTES);
        v16bf aF[2], bF[4];
#pragma unroll
        for (int i = 0; i < 2; ++i) {
            const int st = wm * 2 + i;
            union { v16bf v; bf16x8 h[2]; } u;
            u.h[0] = P[(st * 2 + 0) * 32 + lane];
            u.h[1] = P[(st * 2 + 1) * 32 + lane];
            aF[i] = u.v;
        }
#pragma unroll
        for (int j = 0; j < 4; ++j) {
            const int st = wn * 4 + j;
            union { v16bf v; bf16x8 h[2]; } u;
            u.h[0] = P[B_OFF_B128 + (st * 2 + 0) * 32 + lane];
            u.h[1] = P[B_OFF_B128 + (st * 2 + 1) * 32 + lane];
            bF[j] = u.v;
        }
#pragma unroll
        for (int i = 0; i < 2; ++i)
#pragma unroll
            for (int j = 0; j < 4; ++j)
                acc[i][j] = wmma_bf16(aF[i], bF[j], acc[i][j]);
    };

    // --- prologue: two tile-DMAs in flight (NK >= 8 always here) ---
    stage(0, 0);
    stage(1, 32);

    // --- steady state ---
    for (int s = 0; s < NK - 1; ++s) {
        async_wait<5>();        // stage s complete; stage s+1 may still fly
        __syncthreads();        // cross-wave: everyone's stage s is visible
        compute(s);
        if (s + 2 < NK) stage((s + 2) % 3, (s + 2) << 5);
    }
    // --- final step (nothing left in flight to allow) ---
    async_wait<0>();
    __syncthreads();
    compute(NK - 1);
}

// ---------------------------------------------------------------------------
// GEMM:  C[M,N] = act(A @ Bt^T + bias)
// ---------------------------------------------------------------------------
template<int ACT_GELU, int WRITE_BF, int WRITE_F32>
__global__ __launch_bounds__(256)
void vq_gemm_kernel(const bf16* __restrict__ A, const bf16* __restrict__ Bt,
                    const float* __restrict__ bias,
                    bf16* __restrict__ Cbf, float* __restrict__ Cf,
                    int M, int Nn, int K)
{
    __shared__ char smem[3 * TILE_BYTES];
    const int m0blk = blockIdx.y * 64;
    const int n0blk = blockIdx.x * 256;

    v8f acc[2][4] = {};
    gemm_core_lds(A, Bt, K, m0blk, n0blk, smem, acc);

    const int wave = threadIdx.x >> 5;
    const int m0   = m0blk + (wave >> 2) * 32;
    const int n0   = n0blk + (wave & 3) * 64;
    const int lane = threadIdx.x & 31;
    const int half = lane >> 4;
    const int col  = lane & 15;
#pragma unroll
    for (int i = 0; i < 2; ++i) {
        const int mbase = m0 + 16 * i + half * 8;
#pragma unroll
        for (int j = 0; j < 4; ++j) {
            const int n = n0 + 16 * j + col;
            const float bv = bias[n];
#pragma unroll
            for (int r = 0; r < 8; ++r) {
                float v = acc[i][j][r] + bv;
                if (ACT_GELU) v = gelu_exact(v);
                const size_t off = (size_t)(mbase + r) * Nn + n;
                if (WRITE_BF)  Cbf[off] = (bf16)v;
                if (WRITE_F32) Cf[off]  = v;
            }
        }
    }
}

// ---------------------------------------------------------------------------
// VQ scoring: argmin_n (|e_n|^2 - 2 z.e_n) folded into the GEMM epilogue via
// order-preserving u64 keys + cross-lane min + global atomic_min_u64.
// ---------------------------------------------------------------------------
__global__ __launch_bounds__(256)
void vq_score_kernel(const bf16* __restrict__ Zbf, const bf16* __restrict__ Ebf,
                     const float* __restrict__ enorm,
                     unsigned long long* __restrict__ keys)
{
    __shared__ char smem[3 * TILE_BYTES];
    const int m0blk = blockIdx.y * 64;
    const int n0blk = blockIdx.x * 256;

    v8f acc[2][4] = {};
    gemm_core_lds(Zbf, Ebf, LAT, m0blk, n0blk, smem, acc);

    const int wave = threadIdx.x >> 5;
    const int m0   = m0blk + (wave >> 2) * 32;
    const int n0   = n0blk + (wave & 3) * 64;
    const int lane = threadIdx.x & 31;
    const int half = lane >> 4;
    const int col  = lane & 15;
#pragma unroll
    for (int i = 0; i < 2; ++i) {
        unsigned long long best[8];
#pragma unroll
        for (int r = 0; r < 8; ++r) best[r] = ~0ull;
#pragma unroll
        for (int j = 0; j < 4; ++j) {
            const int n    = n0 + 16 * j + col;
            const float en = enorm[n];
#pragma unroll
            for (int r = 0; r < 8; ++r) {
                float d = en - 2.0f * acc[i][j][r];   // row-constant |z|^2 dropped
                unsigned u = __float_as_uint(d);
                u = (u & 0x80000000u) ? ~u : (u | 0x80000000u);  // ascending map
                unsigned long long key = ((unsigned long long)u << 32) | (unsigned)n;
                if (key < best[r]) best[r] = key;
            }
        }
        const int mbase = m0 + 16 * i + half * 8;
#pragma unroll
        for (int r = 0; r < 8; ++r) {
            unsigned long long k = best[r];
#pragma unroll
            for (int s = 1; s < 16; s <<= 1) {
                unsigned long long o = __shfl_xor(k, s, 32);
                if (o < k) k = o;
            }
            if (col == 0) atomicMin(&keys[mbase + r], k);
        }
    }
}

// ---------------------------------------------------------------------------
// Utility kernels
// ---------------------------------------------------------------------------
__global__ void vq_cvt_kernel(const float* __restrict__ in, bf16* __restrict__ out, size_t n) {
    size_t i = (size_t)blockIdx.x * blockDim.x + threadIdx.x;
    if (i < n) out[i] = (bf16)in[i];
}

// in: [K][Nn] fp32 row-major  ->  out: [Nn][K] bf16   (tiled, coalesced both sides)
__global__ __launch_bounds__(256)
void vq_cvt_transpose_kernel(const float* __restrict__ in, bf16* __restrict__ out,
                             int K, int Nn) {
    __shared__ float tile[32][33];
    const int kb = blockIdx.y * 32;
    const int nb = blockIdx.x * 32;
    const int tx = threadIdx.x & 31;
    const int ty = threadIdx.x >> 5;      // 0..7
#pragma unroll
    for (int r = ty; r < 32; r += 8)
        tile[r][tx] = in[(size_t)(kb + r) * Nn + nb + tx];
    __syncthreads();
#pragma unroll
    for (int r = ty; r < 32; r += 8)
        out[(size_t)(nb + r) * K + kb + tx] = (bf16)tile[tx][r];
}

__global__ void vq_enorm_kernel(const float* __restrict__ emb, float* __restrict__ enorm) {
    int n = blockIdx.x * blockDim.x + threadIdx.x;
    if (n < CODES) {
        float s = 0.f;
        const float* row = emb + (size_t)n * LAT;
        for (int l = 0; l < LAT; ++l) { float v = row[l]; s += v * v; }
        enorm[n] = s;
    }
}

__global__ void vq_init_kernel(unsigned long long* __restrict__ keys, float* __restrict__ loss) {
    size_t i = (size_t)blockIdx.x * blockDim.x + threadIdx.x;
    if (i < NROWS) keys[i] = ~0ull;
    if (i == 0) *loss = 0.f;
}

// one block per row: gather z_q, write index (as float), accumulate squared error
__global__ __launch_bounds__(256)
void vq_gather_loss_kernel(const unsigned long long* __restrict__ keys,
                           const float* __restrict__ emb,
                           const float* __restrict__ z_e,
                           bf16* __restrict__ zq_bf,
                           float* __restrict__ idx_out,
                           float* __restrict__ loss_acc)
{
    const int row = blockIdx.x;
    const int l   = threadIdx.x;
    const int idx = (int)(keys[row] & 0xFFFFFFFFull);
    const float e = emb[(size_t)idx * LAT + l];
    const float z = z_e[(size_t)row * LAT + l];
    zq_bf[(size_t)row * LAT + l] = (bf16)e;
    float d = z - e;
    d *= d;
    __shared__ float red[256];
    red[l] = d;
    __syncthreads();
    for (int s = 128; s > 0; s >>= 1) {
        if (l < s) red[l] += red[l + s];
        __syncthreads();
    }
    if (l == 0) {
        atomicAdd(loss_acc, red[0]);
        idx_out[row] = (float)idx;
    }
}

__global__ void vq_finalize_kernel(const float* __restrict__ loss_acc, float* __restrict__ out) {
    if (threadIdx.x == 0 && blockIdx.x == 0)
        out[0] = 1.25f * (*loss_acc) / (float)((size_t)NROWS * LAT);
}

// ---------------------------------------------------------------------------
// Host-side orchestration
// ---------------------------------------------------------------------------
static inline size_t ws_align(size_t x) { return (x + 255) & ~(size_t)255; }

extern "C" void kernel_launch(void* const* d_in, const int* in_sizes, int n_in,
                              void* d_out, int out_size, void* d_ws, size_t ws_size,
                              hipStream_t stream)
{
    (void)in_sizes; (void)n_in; (void)out_size; (void)ws_size;

    const float* x      = (const float*)d_in[0];
    const float* enc_w1 = (const float*)d_in[1];
    const float* enc_b1 = (const float*)d_in[2];
    const float* enc_w2 = (const float*)d_in[3];
    const float* enc_b2 = (const float*)d_in[4];
    const float* enc_w3 = (const float*)d_in[5];
    const float* enc_b3 = (const float*)d_in[6];
    const float* emb    = (const float*)d_in[7];
    const float* dec_w1 = (const float*)d_in[8];
    const float* dec_b1 = (const float*)d_in[9];
    const float* dec_w2 = (const float*)d_in[10];
    const float* dec_b2 = (const float*)d_in[11];
    const float* dec_w3 = (const float*)d_in[12];
    const float* dec_b3 = (const float*)d_in[13];

    float* out = (float*)d_out;
    const size_t XOUT = (size_t)NROWS * INDIM;         // 8388608
    float* out_loss = out + XOUT;
    float* out_idx  = out + XOUT + 1;

    // ---- carve workspace -------------------------------------------------
    char* ws = (char*)d_ws;
    size_t cur = 0;
    auto carve = [&](size_t bytes) { char* p = ws + cur; cur += ws_align(bytes); return p; };

    bf16* x_bf  = (bf16*)carve((size_t)NROWS * INDIM * 2);
    bf16* w1t   = (bf16*)carve((size_t)HID   * INDIM * 2);   // [HID][INDIM]
    bf16* w2t   = (bf16*)carve((size_t)HID   * HID   * 2);   // [HID][HID]
    bf16* w3t   = (bf16*)carve((size_t)LAT   * HID   * 2);   // [LAT][HID]
    bf16* e_bf  = (bf16*)carve((size_t)CODES * LAT   * 2);   // emb already [CODES][LAT]
    bf16* dw1t  = (bf16*)carve((size_t)HID   * LAT   * 2);   // [HID][LAT]
    bf16* dw2t  = (bf16*)carve((size_t)HID   * HID   * 2);   // [HID][HID]
    bf16* dw3t  = (bf16*)carve((size_t)INDIM * HID   * 2);   // [INDIM][HID]
    bf16* hA    = (bf16*)carve((size_t)NROWS * HID   * 2);
    bf16* hB    = (bf16*)carve((size_t)NROWS * HID   * 2);
    float* z_e  = (float*)carve((size_t)NROWS * LAT  * 4);
    bf16* z_bf  = (bf16*)carve((size_t)NROWS * LAT   * 2);
    bf16* zq_bf = (bf16*)carve((size_t)NROWS * LAT   * 2);
    float* enorm = (float*)carve((size_t)CODES * 4);
    unsigned long long* keys = (unsigned long long*)carve((size_t)NROWS * 8);
    float* loss_acc = (float*)carve(256);

    const int TPB = 256;
    auto blocks = [&](size_t n) { return (unsigned)((n + TPB - 1) / TPB); };
    auto tgrid  = [&](int K, int Nn) { return dim3(Nn / 32, K / 32); };

    // ---- conversions -----------------------------------------------------
    vq_cvt_kernel<<<blocks((size_t)NROWS * INDIM), TPB, 0, stream>>>(x, x_bf, (size_t)NROWS * INDIM);
    vq_cvt_transpose_kernel<<<tgrid(INDIM, HID), TPB, 0, stream>>>(enc_w1, w1t, INDIM, HID);
    vq_cvt_transpose_kernel<<<tgrid(HID, HID),   TPB, 0, stream>>>(enc_w2, w2t, HID, HID);
    vq_cvt_transpose_kernel<<<tgrid(HID, LAT),   TPB, 0, stream>>>(enc_w3, w3t, HID, LAT);
    vq_cvt_kernel<<<blocks((size_t)CODES * LAT), TPB, 0, stream>>>(emb, e_bf, (size_t)CODES * LAT);
    vq_cvt_transpose_kernel<<<tgrid(LAT, HID),   TPB, 0, stream>>>(dec_w1, dw1t, LAT, HID);
    vq_cvt_transpose_kernel<<<tgrid(HID, HID),   TPB, 0, stream>>>(dec_w2, dw2t, HID, HID);
    vq_cvt_transpose_kernel<<<tgrid(HID, INDIM), TPB, 0, stream>>>(dec_w3, dw3t, HID, INDIM);
    vq_enorm_kernel<<<blocks(CODES), TPB, 0, stream>>>(emb, enorm);
    vq_init_kernel<<<blocks(NROWS), TPB, 0, stream>>>(keys, loss_acc);

    // ---- encoder ---------------------------------------------------------
    vq_gemm_kernel<1, 1, 0><<<dim3(HID / 256, NROWS / 64), TPB, 0, stream>>>(
        x_bf, w1t, enc_b1, hA, nullptr, NROWS, HID, INDIM);
    vq_gemm_kernel<1, 1, 0><<<dim3(HID / 256, NROWS / 64), TPB, 0, stream>>>(
        hA, w2t, enc_b2, hB, nullptr, NROWS, HID, HID);
    vq_gemm_kernel<0, 1, 1><<<dim3(LAT / 256, NROWS / 64), TPB, 0, stream>>>(
        hB, w3t, enc_b3, z_bf, z_e, NROWS, LAT, HID);

    // ---- vector quantization --------------------------------------------
    vq_score_kernel<<<dim3(CODES / 256, NROWS / 64), TPB, 0, stream>>>(z_bf, e_bf, enorm, keys);
    vq_gather_loss_kernel<<<NROWS, LAT, 0, stream>>>(keys, emb, z_e, zq_bf, out_idx, loss_acc);
    vq_finalize_kernel<<<1, 32, 0, stream>>>(loss_acc, out_loss);

    // ---- decoder (z_out == z_q numerically) ------------------------------
    vq_gemm_kernel<1, 1, 0><<<dim3(HID / 256, NROWS / 64), TPB, 0, stream>>>(
        zq_bf, dw1t, dec_b1, hA, nullptr, NROWS, HID, LAT);
    vq_gemm_kernel<1, 1, 0><<<dim3(HID / 256, NROWS / 64), TPB, 0, stream>>>(
        hA, dw2t, dec_b2, hB, nullptr, NROWS, HID, HID);
    vq_gemm_kernel<0, 0, 1><<<dim3(INDIM / 256, NROWS / 64), TPB, 0, stream>>>(
        hB, dw3t, dec_b3, nullptr, out, NROWS, INDIM, HID);
}
